// MultiheadAttentionSep_29523605193087
// MI455X (gfx1250) — matrix operand order repaired and checked by reading, not compiled, not run on hardware
//
#include <hip/hip_runtime.h>
#include <hip/hip_bf16.h>
#include <stdint.h>

#define B_  8
#define LV_ 1024
#define LL_ 1024
#define LO_ 512
#define E_  768
#define H_  12
#define D_  64
#define LS_ (LV_ + LL_)

typedef __attribute__((ext_vector_type(16))) __bf16 v16bf;
typedef __attribute__((ext_vector_type(8)))  float  v8f;

// ---------- fragment helpers (wave32, CDNA5 16x16x32 bf16 layouts) ----------

static __device__ __forceinline__ v16bf pack2(uint4 lo, uint4 hi) {
  union { uint4 u[2]; v16bf v; } t;
  t.u[0] = lo; t.u[1] = hi;
  return t.v;
}

// A-matrix 16x32 (MxK) from bf16 row-major, leading dim ld (elements).
// Lanes 0-15: row=lane, K={0..7,16..23}; lanes 16-31: row=lane-16, K={8..15,24..31}.
static __device__ __forceinline__ v16bf load_a_bf16(const __bf16* src, int ld) {
  const int lane = threadIdx.x & 31;
  const int row  = lane & 15;
  const int k0   = (lane < 16) ? 0 : 8;
  const __bf16* p = src + (size_t)row * ld + k0;
  return pack2(*(const uint4*)p, *(const uint4*)(p + 16));
}

// B-matrix 32x16 (KxN) as W^T, W row-major [N,K] bf16, leading dim ld.
// Lanes 0-15: col=lane, K={0..7,16..23}; lanes 16-31: col=lane-16, K={8..15,24..31}.
static __device__ __forceinline__ v16bf load_bT_bf16(const __bf16* w, int ld) {
  const int lane = threadIdx.x & 31;
  const int col  = lane & 15;
  const int k0   = (lane < 16) ? 0 : 8;
  const __bf16* p = w + (size_t)col * ld + k0;
  return pack2(*(const uint4*)p, *(const uint4*)(p + 16));
}

static __device__ __forceinline__ v8f wmma_bf16(v16bf a, v16bf b, v8f c) {
  return __builtin_amdgcn_wmma_f32_16x16x32_bf16(false, a, false, b, (short)0, c, false, false);
}

// ---------- conversion / init kernels ----------

__global__ void zero_f32(float* __restrict__ p, int n) {
  int i = blockIdx.x * blockDim.x + threadIdx.x;
  if (i < n) p[i] = 0.0f;
}

__global__ void cvt_f32_bf16(const float* __restrict__ s, __bf16* __restrict__ d, int n4) {
  int i = blockIdx.x * blockDim.x + threadIdx.x;
  if (i < n4) {
    float4 f = ((const float4*)s)[i];
    union { __bf16 e[4]; uint2 u; } t;
    t.e[0] = (__bf16)f.x; t.e[1] = (__bf16)f.y;
    t.e[2] = (__bf16)f.z; t.e[3] = (__bf16)f.w;
    ((uint2*)d)[i] = t.u;
  }
}

// ---------- GEMM: C[M,N] = A[M,K] @ W[N,K]^T + bias[N] ----------
// Block: 256 thr = 8 waves stacked along M; block tile 256M x 128N; per-wave 32M x 128N.
// 128x32 weight tile staged to LDS with double-buffered async copies; every B fragment
// is reused by all 8 waves. 16 WMMAs per wave per K=32 step.
// mode 0: bf16 out row-major; mode 1: bf16 out as [b][h][d][s]; mode 2: f32 out row-major.
__global__ __launch_bounds__(256) void gemm_bias_wmma(
    const __bf16* __restrict__ A, const __bf16* __restrict__ W,
    const float* __restrict__ bias, void* __restrict__ outp,
    int N, int K, int mode, int Lseq) {
  __shared__ __bf16 Wt[2][128 * 32];  // 2 x 8 KB double buffer

  const int tid  = threadIdx.x;
  const int wave = tid >> 5;
  const int lane = tid & 31;
  const int bn0  = blockIdx.x * 128;
  const int m0   = blockIdx.y * 256 + wave * 32;

  // each thread asyncs 32 B of the 128x32 weight tile (row-major, 32 bf16/row)
  const int prow = tid >> 1;          // 0..127
  const int pcol = (tid & 1) * 16;    // element offset in row

  const int NK = K / 32;

  // prefetch k-chunk 0 into buffer 0
  {
    const __bf16* g = W + (size_t)(bn0 + prow) * K + pcol;
    uint32_t l = (uint32_t)(uintptr_t)&Wt[0][prow * 32 + pcol];
    asm volatile("global_load_async_to_lds_b128 %0, %1, off\n\t"
                 "global_load_async_to_lds_b128 %0, %1, off offset:16"
                 :: "v"(l), "v"(g) : "memory");
  }

  v8f acc[2][8] = {};
  for (int ks = 0; ks < NK; ++ks) {
    if (ks + 1 < NK) {
      const __bf16* g = W + (size_t)(bn0 + prow) * K + (ks + 1) * 32 + pcol;
      uint32_t l = (uint32_t)(uintptr_t)&Wt[(ks + 1) & 1][prow * 32 + pcol];
      asm volatile("global_load_async_to_lds_b128 %0, %1, off\n\t"
                   "global_load_async_to_lds_b128 %0, %1, off offset:16"
                   :: "v"(l), "v"(g) : "memory");
      asm volatile("s_wait_asynccnt 0x2" ::: "memory");  // current chunk landed
    } else {
      asm volatile("s_wait_asynccnt 0x0" ::: "memory");
    }
    __syncthreads();

    const int k = ks * 32;
    const v16bf a0 = load_a_bf16(A + (size_t)m0 * K + k, K);
    const v16bf a1 = load_a_bf16(A + (size_t)(m0 + 16) * K + k, K);
    const __bf16* wb = &Wt[ks & 1][0];

    // load all 8 B fragments first: one dscnt wait, 16 independent WMMAs after
    v16bf bf[8];
#pragma unroll
    for (int j = 0; j < 8; ++j) bf[j] = load_bT_bf16(wb + (size_t)(j * 16) * 32, 32);
#pragma unroll
    for (int j = 0; j < 8; ++j) {
      acc[0][j] = wmma_bf16(a0, bf[j], acc[0][j]);
      acc[1][j] = wmma_bf16(a1, bf[j], acc[1][j]);
    }
    __syncthreads();  // reads done before this buffer is overwritten
  }

  const int col = lane & 15;
  const int rb  = (lane < 16) ? 0 : 8;
#pragma unroll
  for (int mi = 0; mi < 2; ++mi) {
#pragma unroll
    for (int j = 0; j < 8; ++j) {
      const int n = bn0 + j * 16 + col;
      const float bj = bias[n];
#pragma unroll
      for (int v = 0; v < 8; ++v) {
        const int m = m0 + mi * 16 + rb + v;
        const float val = acc[mi][j][v] + bj;
        if (mode == 2) {
          ((float*)outp)[(size_t)m * N + n] = val;
        } else if (mode == 0) {
          ((__bf16*)outp)[(size_t)m * N + n] = (__bf16)val;
        } else {
          const int bb = m / Lseq;
          const int s  = m - bb * Lseq;
          const int h  = n >> 6;
          const int dd = n & 63;
          ((__bf16*)outp)[(((size_t)bb * H_ + h) * D_ + dd) * Lseq + s] = (__bf16)val;
        }
      }
    }
  }
}

// ---------- fused dual-source attention ----------
// One block (4 waves) per (b, h, 16-query tile). Full-row softmax in LDS.
__global__ __launch_bounds__(128) void attn_fused(
    const __bf16* __restrict__ q1, const __bf16* __restrict__ q2,
    const __bf16* __restrict__ k1, const __bf16* __restrict__ k2,
    const __bf16* __restrict__ v1, const __bf16* __restrict__ v2,
    const float* __restrict__ mask, __bf16* __restrict__ ctx,
    float* __restrict__ attn_avg) {
  __shared__ __bf16 P[16 * LS_];            // 64 KB: scores -> probs
  __shared__ float  ctxred[4 * 16 * D_];    // 16 KB: per-wave ctx partials
  __shared__ float  red[16][8];
  __shared__ float  rowstat[16];

  const int b    = blockIdx.z;
  const int h    = blockIdx.y;
  const int t0   = blockIdx.x * 16;
  const int wave = threadIdx.x >> 5;
  const int lane = threadIdx.x & 31;
  const float scale = 0.125f;  // 1/sqrt(64)

  const __bf16* q1p = q1 + ((size_t)b * LO_ + t0) * E_ + h * D_;
  const __bf16* q2p = q2 + ((size_t)b * LO_ + t0) * E_ + h * D_;
  const v16bf qa10 = load_a_bf16(q1p, E_);
  const v16bf qa11 = load_a_bf16(q1p + 32, E_);
  const v16bf qa20 = load_a_bf16(q2p, E_);
  const v16bf qa21 = load_a_bf16(q2p + 32, E_);

  const int col = lane & 15;
  const int rb  = (lane < 16) ? 0 : 8;

  // ---- scores: S[t, s] = q.k * scale + mask ----
  for (int kt = wave; kt < LS_ / 16; kt += 4) {
    const int s0 = kt * 16;
    const bool fromV = (s0 < LV_);
    const __bf16* kp = fromV
        ? (k1 + ((size_t)b * LV_ + s0) * E_ + h * D_)
        : (k2 + ((size_t)b * LL_ + (s0 - LV_)) * E_ + h * D_);
    const v16bf kb0 = load_bT_bf16(kp, E_);
    const v16bf kb1 = load_bT_bf16(kp + 32, E_);
    v8f c = {};
    c = wmma_bf16(fromV ? qa10 : qa20, kb0, c);
    c = wmma_bf16(fromV ? qa11 : qa21, kb1, c);
#pragma unroll
    for (int v = 0; v < 8; ++v) {
      const int m  = rb + v;
      const int sg = s0 + col;
      const float sc = c[v] * scale + mask[(size_t)(t0 + m) * LS_ + sg];
      P[m * LS_ + sg] = (__bf16)sc;
    }
  }
  __syncthreads();

  // ---- softmax: 8 threads per row, each owns a contiguous 256-element span ----
  const int row = threadIdx.x >> 3;
  const int sub = threadIdx.x & 7;
  const int base = row * LS_ + sub * 256;

  float mx = -3.0e38f;
  for (int c = 0; c < 256; c += 8) {
    union { uint4 u; __bf16 e[8]; } t;
    t.u = *(const uint4*)&P[base + c];
#pragma unroll
    for (int i = 0; i < 8; ++i) mx = fmaxf(mx, (float)t.e[i]);
  }
  red[row][sub] = mx;
  __syncthreads();
  if (sub == 0) {
    float m2 = red[row][0];
#pragma unroll
    for (int i = 1; i < 8; ++i) m2 = fmaxf(m2, red[row][i]);
    rowstat[row] = m2;
  }
  __syncthreads();
  const float rm = rowstat[row];

  float psum = 0.0f;
  for (int c = 0; c < 256; c += 8) {
    union { uint4 u; __bf16 e[8]; } t;
    t.u = *(const uint4*)&P[base + c];
#pragma unroll
    for (int i = 0; i < 8; ++i) {
      const float e = __expf((float)t.e[i] - rm);
      t.e[i] = (__bf16)e;
      psum += e;
    }
    *(uint4*)&P[base + c] = t.u;
  }
  red[row][sub] = psum;
  __syncthreads();
  if (sub == 0) {
    float s2 = 0.0f;
#pragma unroll
    for (int i = 0; i < 8; ++i) s2 += red[row][i];
    rowstat[row] = s2;
  }
  __syncthreads();
  const float inv = 1.0f / rowstat[row];
  const float hscale = inv * (1.0f / (float)H_);
  float* avg_row = attn_avg + ((size_t)b * LO_ + t0 + row) * LS_ + sub * 256;

  for (int c = 0; c < 256; c += 8) {
    union { uint4 u; __bf16 e[8]; } t;
    t.u = *(const uint4*)&P[base + c];
#pragma unroll
    for (int i = 0; i < 8; ++i) {
      const float e = (float)t.e[i];
      t.e[i] = (__bf16)(e * inv);
      atomicAdd(&avg_row[c + i], e * hscale);  // mean over heads across blocks
    }
    *(uint4*)&P[base + c] = t.u;
  }
  __syncthreads();

  // ---- ctx = P[16x2048] @ Vcat[2048x64]; per-wave K partials ----
  v8f acc[4] = {};
  for (int kc = wave; kc < LS_ / 32; kc += 4) {
    const int s0 = kc * 32;
    const bool fromV = (s0 < LV_);
    const __bf16* vb = fromV
        ? (v1 + (((size_t)b * H_ + h) * D_) * (size_t)LV_ + s0)
        : (v2 + (((size_t)b * H_ + h) * D_) * (size_t)LL_ + (s0 - LV_));
    const int vld = LV_;  // == LL_
    const v16bf pa = load_a_bf16(P + s0, LS_);
    v16bf vf[4];
#pragma unroll
    for (int j = 0; j < 4; ++j) vf[j] = load_bT_bf16(vb + (size_t)(j * 16) * vld, vld);
#pragma unroll
    for (int j = 0; j < 4; ++j) acc[j] = wmma_bf16(pa, vf[j], acc[j]);
  }
  __syncthreads();
#pragma unroll
  for (int j = 0; j < 4; ++j) {
#pragma unroll
    for (int v = 0; v < 8; ++v) {
      ctxred[wave * (16 * D_) + (rb + v) * D_ + j * 16 + col] = acc[j][v];
    }
  }
  __syncthreads();
  for (int idx = threadIdx.x; idx < 16 * D_; idx += 128) {
    const float s = ctxred[idx] + ctxred[idx + 16 * D_] +
                    ctxred[idx + 2 * 16 * D_] + ctxred[idx + 3 * 16 * D_];
    const int m = idx >> 6;
    const int n = idx & 63;
    ctx[((size_t)b * LO_ + t0 + m) * E_ + h * D_ + n] = (__bf16)s;
  }
}

// ---------- host launcher ----------

extern "C" void kernel_launch(void* const* d_in, const int* in_sizes, int n_in,
                              void* d_out, int out_size, void* d_ws, size_t ws_size,
                              hipStream_t stream) {
  (void)in_sizes; (void)n_in; (void)out_size; (void)ws_size;

  const float* V  = (const float*)d_in[0];
  const float* L  = (const float*)d_in[1];
  const float* O  = (const float*)d_in[2];
  const float* mk = (const float*)d_in[3];
  const float* w1 = (const float*)d_in[4];
  const float* b1 = (const float*)d_in[5];
  const float* w2 = (const float*)d_in[6];
  const float* b2 = (const float*)d_in[7];
  const float* ow = (const float*)d_in[8];
  const float* ob = (const float*)d_in[9];

  float* out      = (float*)d_out;
  float* attn_avg = out + (size_t)B_ * LO_ * E_;

  __bf16* ws = (__bf16*)d_ws;
  size_t o = 0;
  __bf16* Vb   = ws + o; o += (size_t)B_ * LV_ * E_;
  __bf16* Lb   = ws + o; o += (size_t)B_ * LL_ * E_;
  __bf16* Ob   = ws + o; o += (size_t)B_ * LO_ * E_;
  __bf16* w1b  = ws + o; o += (size_t)3 * E_ * E_;
  __bf16* w2b  = ws + o; o += (size_t)3 * E_ * E_;
  __bf16* owb  = ws + o; o += (size_t)E_ * E_;
  __bf16* q1   = ws + o; o += (size_t)B_ * LO_ * E_;
  __bf16* q2   = ws + o; o += (size_t)B_ * LO_ * E_;
  __bf16* k1   = ws + o; o += (size_t)B_ * LV_ * E_;
  __bf16* k2   = ws + o; o += (size_t)B_ * LL_ * E_;
  __bf16* v1   = ws + o; o += (size_t)B_ * LV_ * E_;
  __bf16* v2   = ws + o; o += (size_t)B_ * LL_ * E_;
  __bf16* ctxb = ws + o; o += (size_t)B_ * LO_ * E_;

  // attn_avg accumulates via atomics -> zero it first
  {
    const int n = B_ * LO_ * LS_;
    zero_f32<<<(n + 255) / 256, 256, 0, stream>>>(attn_avg, n);
  }

  // f32 -> bf16 conversions (inputs + weights)
  const dim3 cb(256);
  auto cvt = [&](const float* s, __bf16* d, size_t n) {
    int n4 = (int)(n / 4);
    cvt_f32_bf16<<<(n4 + 255) / 256, cb, 0, stream>>>(s, d, n4);
  };
  cvt(V,  Vb,  (size_t)B_ * LV_ * E_);
  cvt(L,  Lb,  (size_t)B_ * LL_ * E_);
  cvt(O,  Ob,  (size_t)B_ * LO_ * E_);
  cvt(w1, w1b, (size_t)3 * E_ * E_);
  cvt(w2, w2b, (size_t)3 * E_ * E_);
  cvt(ow, owb, (size_t)E_ * E_);

  const dim3 blk(256);
  const dim3 gq(E_ / 128, (B_ * LO_) / 256);  // 6 x 16
  const dim3 gk(E_ / 128, (B_ * LV_) / 256);  // 6 x 32
  const size_t WS = (size_t)E_ * E_;

  // projections (q/k row-major bf16 [b,t,E]; v transposed bf16 [b,h,d,s])
  gemm_bias_wmma<<<gq, blk, 0, stream>>>(Ob, w1b,          b1,          q1, E_, E_, 0, 1);
  gemm_bias_wmma<<<gq, blk, 0, stream>>>(Ob, w2b,          b2,          q2, E_, E_, 0, 1);
  gemm_bias_wmma<<<gk, blk, 0, stream>>>(Vb, w1b + WS,     b1 + E_,     k1, E_, E_, 0, 1);
  gemm_bias_wmma<<<gk, blk, 0, stream>>>(Lb, w2b + WS,     b2 + E_,     k2, E_, E_, 0, 1);
  gemm_bias_wmma<<<gk, blk, 0, stream>>>(Vb, w1b + 2 * WS, b1 + 2 * E_, v1, E_, E_, 1, LV_);
  gemm_bias_wmma<<<gk, blk, 0, stream>>>(Lb, w2b + 2 * WS, b2 + 2 * E_, v2, E_, E_, 1, LL_);

  // fused dual-source attention
  const dim3 ga(LO_ / 16, H_, B_);
  attn_fused<<<ga, dim3(128), 0, stream>>>(q1, q2, k1, k2, v1, v2, mk, ctxb, attn_avg);

  // output projection (f32 out)
  gemm_bias_wmma<<<gq, blk, 0, stream>>>(ctxb, owb, ob, out, E_, E_, 2, 1);
}
